// NodeClassificationGNN_22754736734765
// MI455X (gfx1250) — matrix operand order repaired and checked by reading, not compiled, not run on hardware
//
#include <hip/hip_runtime.h>
#include <hip/hip_bf16.h>

#define N_NODES 2048
#define N_FEAT  512
#define N_CLS   128
#define K_TOT   2560   // N_FEAT + N_NODES
#define INF_F   1e9f

#define GRID_BF 32
#define COLS_PB (N_NODES / GRID_BF)          // 64 columns per block
#define NGROUP  (512 / (COLS_PB / 4))        // 32 u-groups per block
#define U_PER_G (N_NODES / NGROUP)           // 64 u's per group

typedef __attribute__((ext_vector_type(16))) __bf16 v16bf;
typedef __attribute__((ext_vector_type(8)))  __bf16 v8bf;
typedef __attribute__((ext_vector_type(8)))  float  v8f;

union Frag { v8bf h[2]; v16bf v; };

// ---------------- prep kernels ----------------

// emb (2048x512 f32) -> featB[:, 0:512] bf16
__global__ void prep_emb_kernel(const float* __restrict__ emb,
                                __hip_bfloat16* __restrict__ featB) {
    int idx = blockIdx.x * 256 + threadIdx.x;   // 0 .. 2048*512-1
    int v = idx >> 9;
    int k = idx & 511;
    featB[(size_t)v * K_TOT + k] = __float2bfloat16(emb[idx]);
}

// fc_weight (128x2560 f32) -> WB bf16
__global__ void prep_w_kernel(const float* __restrict__ W,
                              __hip_bfloat16* __restrict__ WB) {
    int idx = blockIdx.x * 256 + threadIdx.x;   // 0 .. 128*2560-1
    WB[idx] = __float2bfloat16(W[idx]);
}

// dist_0[v] = (v==src) ? 0 : INF ; featB[:, 512] = bf16(dist_0) ; flag/barrier = 0
__global__ void prep_dist_kernel(const int* __restrict__ src,
                                 float* __restrict__ dist0,
                                 __hip_bfloat16* __restrict__ featB,
                                 float* __restrict__ flag,
                                 unsigned* __restrict__ bar) {
    int v = blockIdx.x * 256 + threadIdx.x;     // 0 .. 2047
    float d = (v == src[0]) ? 0.0f : INF_F;
    dist0[v] = d;
    featB[(size_t)v * K_TOT + N_FEAT] = __float2bfloat16(d);
    if (v == 0) {
        flag[0] = 0.0f;
        bar[0] = 0u;   // arrival counter
        bar[1] = 0u;   // generation
    }
}

// ---------------- device-wide barrier (sense via generation counter) ----------------
__device__ __forceinline__ void grid_sync(unsigned* bar_cnt, unsigned* bar_gen,
                                          unsigned nblocks) {
    __syncthreads();
    if (threadIdx.x == 0) {
        unsigned g = __hip_atomic_load(bar_gen, __ATOMIC_RELAXED,
                                       __HIP_MEMORY_SCOPE_AGENT);
        unsigned a = __hip_atomic_fetch_add(bar_cnt, 1u, __ATOMIC_ACQ_REL,
                                            __HIP_MEMORY_SCOPE_AGENT);
        if (a == nblocks - 1) {
            __hip_atomic_store(bar_cnt, 0u, __ATOMIC_RELAXED,
                               __HIP_MEMORY_SCOPE_AGENT);
            __hip_atomic_store(bar_gen, g + 1u, __ATOMIC_RELEASE,
                               __HIP_MEMORY_SCOPE_AGENT);
        } else {
            while (__hip_atomic_load(bar_gen, __ATOMIC_ACQUIRE,
                                     __HIP_MEMORY_SCOPE_AGENT) == g) {
                __builtin_amdgcn_s_sleep(1);
            }
        }
    }
    __syncthreads();
}

// ---------------- persistent Bellman-Ford ----------------
// All 2047 relaxation sweeps + final negative-cycle round in ONE kernel.
// 32 blocks x 512 threads; block owns 64 columns, 32 u-groups of 64 each.
// dist vector staged in LDS (broadcast); partials reduced across groups in LDS;
// sweeps separated by an agent-scope grid barrier.
__global__ __launch_bounds__(512)
void bf_persistent_kernel(const float* __restrict__ adj,
                          float* __restrict__ buf0,
                          float* __restrict__ buf1,
                          __hip_bfloat16* __restrict__ featB,
                          float* __restrict__ flag,
                          unsigned* __restrict__ bar) {
    __shared__ float sdist[N_NODES];              // 8 KB
    __shared__ float spart[NGROUP * COLS_PB];     // 32*64 = 2048 floats, 8 KB

    const int tid  = threadIdx.x;
    const int q    = tid & (COLS_PB / 4 - 1);     // 16 quads -> 64 columns
    const int g    = tid >> 4;                    // 32 u-groups
    const int vloc = q * 4;
    const int v    = blockIdx.x * COLS_PB + vloc;
    const int u0   = g * U_PER_G;

    for (int iter = 1; iter <= N_NODES; ++iter) {
        const float* din  = (iter & 1) ? buf0 : buf1;
        float*       dout = (iter & 1) ? buf1 : buf0;

        for (int u = tid; u < N_NODES; u += 512) sdist[u] = din[u];
        __syncthreads();

        float4 m = make_float4(INF_F, INF_F, INF_F, INF_F);
        #pragma unroll 4
        for (int uu = 0; uu < U_PER_G; ++uu) {
            int u = u0 + uu;
            float du = sdist[u];
            float4 a = reinterpret_cast<const float4*>(adj + (size_t)u * N_NODES + v)[0];
            m.x = fminf(m.x, du + a.x);
            m.y = fminf(m.y, du + a.y);
            m.z = fminf(m.z, du + a.z);
            m.w = fminf(m.w, du + a.w);
        }
        spart[g * COLS_PB + vloc + 0] = m.x;
        spart[g * COLS_PB + vloc + 1] = m.y;
        spart[g * COLS_PB + vloc + 2] = m.z;
        spart[g * COLS_PB + vloc + 3] = m.w;
        __syncthreads();

        if (tid < COLS_PB) {
            float r = spart[tid];
            #pragma unroll
            for (int gg = 1; gg < NGROUP; ++gg) r = fminf(r, spart[gg * COLS_PB + tid]);
            int vv = blockIdx.x * COLS_PB + tid;
            if (iter < N_NODES) {
                r = fminf(r, sdist[vv]);          // new = min(dist, relax)
                dout[vv] = r;
                featB[(size_t)vv * K_TOT + N_FEAT + iter] = __float2bfloat16(r);
            } else {
                // extra relaxation: any strict improvement -> negative cycle
                if (r < sdist[vv]) flag[0] = 1.0f;
            }
        }
        grid_sync(bar, bar + 1, GRID_BF);
    }
}

// ---------------- WMMA GEMM: out = featB(2048x2560,bf16) @ WB^T + bias ----------------
// 256 threads = 8 waves per block, 32 blocks = 256 waves.
// Wave: 4 M-tiles x 1 N-tile; register double-buffered K-loop of 80 WMMAs.
__global__ __launch_bounds__(256)
void gemm_wmma_kernel(const __hip_bfloat16* __restrict__ featB,
                      const __hip_bfloat16* __restrict__ WB,
                      const float* __restrict__ bias,
                      float* __restrict__ out) {
    int warp = threadIdx.x >> 5;
    int lane = threadIdx.x & 31;
    int w      = blockIdx.x * 8 + warp;
    int ntile  = w & 7;          // 8 N-tiles of 16 classes
    int mgroup = w >> 3;         // 0..31, 4 M-tiles each
    int mbase  = mgroup * 64;
    int laneRow = lane & 15;
    int hi      = lane >> 4;     // K-half select

    float bc = bias[ntile * 16 + laneRow];
    v8f acc0 = {bc, bc, bc, bc, bc, bc, bc, bc};
    v8f acc1 = acc0, acc2 = acc0, acc3 = acc0;

    // B fragment: lane holds class column; K runs {hi*8..+7} and {hi*8+16..+23}
    const __hip_bfloat16* pB = WB + (size_t)(ntile * 16 + laneRow) * K_TOT + hi * 8;
    // A fragment: lane holds row; K runs {hi*8..+7} and {hi*8+16..+23}
    const __hip_bfloat16* pA = featB + (size_t)(mbase + laneRow) * K_TOT + hi * 8;
    const size_t rs = (size_t)16 * K_TOT;

#define LOADF(B, A0, A1, A2, A3, KS) do { int kof = (KS) * 32;                         \
        (B).h[0]  = *reinterpret_cast<const v8bf*>(pB + kof);                          \
        (B).h[1]  = *reinterpret_cast<const v8bf*>(pB + kof + 16);                     \
        (A0).h[0] = *reinterpret_cast<const v8bf*>(pA + kof);                          \
        (A0).h[1] = *reinterpret_cast<const v8bf*>(pA + kof + 16);                     \
        (A1).h[0] = *reinterpret_cast<const v8bf*>(pA + rs + kof);                     \
        (A1).h[1] = *reinterpret_cast<const v8bf*>(pA + rs + kof + 16);                \
        (A2).h[0] = *reinterpret_cast<const v8bf*>(pA + 2 * rs + kof);                 \
        (A2).h[1] = *reinterpret_cast<const v8bf*>(pA + 2 * rs + kof + 16);            \
        (A3).h[0] = *reinterpret_cast<const v8bf*>(pA + 3 * rs + kof);                 \
        (A3).h[1] = *reinterpret_cast<const v8bf*>(pA + 3 * rs + kof + 16);            \
    } while (0)

#define DO_WMMA(B, A0, A1, A2, A3)                                                     \
        acc0 = __builtin_amdgcn_wmma_f32_16x16x32_bf16(false, (A0).v, false, (B).v,    \
                                                       (short)0, acc0, false, false);  \
        acc1 = __builtin_amdgcn_wmma_f32_16x16x32_bf16(false, (A1).v, false, (B).v,    \
                                                       (short)0, acc1, false, false);  \
        acc2 = __builtin_amdgcn_wmma_f32_16x16x32_bf16(false, (A2).v, false, (B).v,    \
                                                       (short)0, acc2, false, false);  \
        acc3 = __builtin_amdgcn_wmma_f32_16x16x32_bf16(false, (A3).v, false, (B).v,    \
                                                       (short)0, acc3, false, false)

    Frag b0, a00, a10, a20, a30;   // buffer 0
    Frag b1, a01, a11, a21, a31;   // buffer 1

    LOADF(b0, a00, a10, a20, a30, 0);
    for (int ks = 0; ks < K_TOT / 32; ks += 2) {
        LOADF(b1, a01, a11, a21, a31, ks + 1);     // prefetch odd step
        DO_WMMA(b0, a00, a10, a20, a30);           // consume even step
        if (ks + 2 < K_TOT / 32) {
            LOADF(b0, a00, a10, a20, a30, ks + 2); // prefetch next even step
        }
        DO_WMMA(b1, a01, a11, a21, a31);           // consume odd step
    }
#undef LOADF
#undef DO_WMMA

    int col = ntile * 16 + laneRow;
    #pragma unroll
    for (int i = 0; i < 8; ++i) {
        int row = mbase + i + hi * 8;
        out[(size_t)(row)      * N_CLS + col] = acc0[i];
        out[(size_t)(row + 16) * N_CLS + col] = acc1[i];
        out[(size_t)(row + 32) * N_CLS + col] = acc2[i];
        out[(size_t)(row + 48) * N_CLS + col] = acc3[i];
    }
}

// ---------------- launch ----------------
extern "C" void kernel_launch(void* const* d_in, const int* in_sizes, int n_in,
                              void* d_out, int out_size, void* d_ws, size_t ws_size,
                              hipStream_t stream) {
    const float* adj  = (const float*)d_in[0];   // 2048x2048
    const float* emb  = (const float*)d_in[1];   // 2048x512
    const float* W    = (const float*)d_in[2];   // 128x2560
    const float* bias = (const float*)d_in[3];   // 128
    const int*   src  = (const int*)d_in[4];     // scalar

    float* out  = (float*)d_out;                 // 2048x128 logits
    float* flag = out + (size_t)N_NODES * N_CLS; // has_negative_cycle at [262144]

    char* ws = (char*)d_ws;
    float*    buf0 = (float*)ws;                             // dist ping (8 KB)
    float*    buf1 = buf0 + N_NODES;                         // dist pong (8 KB)
    unsigned* bar  = (unsigned*)(ws + 2 * N_NODES * sizeof(float)); // 16 B barrier state
    __hip_bfloat16* featB = (__hip_bfloat16*)(ws + 2 * N_NODES * sizeof(float) + 16);
    __hip_bfloat16* WB    = featB + (size_t)N_NODES * K_TOT; // after ~10 MB featB

    prep_emb_kernel<<<(N_NODES * N_FEAT) / 256, 256, 0, stream>>>(emb, featB);
    prep_w_kernel<<<(N_CLS * K_TOT) / 256, 256, 0, stream>>>(W, WB);
    prep_dist_kernel<<<N_NODES / 256, 256, 0, stream>>>(src, buf0, featB, flag, bar);

    bf_persistent_kernel<<<GRID_BF, 512, 0, stream>>>(adj, buf0, buf1, featB, flag, bar);

    gemm_wmma_kernel<<<32, 256, 0, stream>>>(featB, WB, bias, out);
}
